// PreEncoderBlockHFTT_66718021976271
// MI455X (gfx1250) — matrix operand order repaired and checked by reading, not compiled, not run on hardware
//
#include <hip/hip_runtime.h>
#include <hip/hip_bf16.h>

typedef __attribute__((ext_vector_type(16))) __bf16       v16bf;
typedef __attribute__((ext_vector_type(8)))  float        v8f;
typedef __attribute__((ext_vector_type(8)))  unsigned int v8u;

#define B_    8
#define T_    512
#define F_    128
#define E_    128
#define KP_   64      // padded tap count (WMMA K = 2 x 32)
#define TAPS_ 36      // effective taps after folding conv into linear
#define PADV  1e-7f

// ---------------------------------------------------------------------------
// Prep: fold the 5-tap conv (4 channels) into the 128x128 linear layer,
// producing W_eff[e][64] (bf16, taps >=36 are zero) and B_eff[e] (f32).
// One block of 128 threads; e = threadIdx.x.
// ---------------------------------------------------------------------------
__global__ __launch_bounds__(128) void prep_kernel(
    const float* __restrict__ conv_w,  // (4,1,1,5)
    const float* __restrict__ conv_b,  // (4,)
    const float* __restrict__ lin_w,   // (128,128)
    const float* __restrict__ lin_b,   // (128,)
    __bf16* __restrict__ Wbf,          // (128,64) bf16
    float*  __restrict__ Beff)         // (128,) f32
{
  const int e = threadIdx.x;

  float bias = lin_b[e];
  for (int o = 0; o < 4; ++o) {
    float s = 0.f;
    for (int w = 0; w < 32; ++w) s += lin_w[e * E_ + o * 32 + w];
    bias += conv_b[o] * s;
  }
  Beff[e] = bias;

  for (int j = 0; j < KP_; ++j) {
    float acc = 0.f;
    if (j < TAPS_) {
      for (int o = 0; o < 4; ++o)
        for (int k = 0; k < 5; ++k) {
          int w = j - k;
          if (w >= 0 && w < 32)
            acc += conv_w[o * 5 + k] * lin_w[e * E_ + o * 32 + w];
        }
    }
    Wbf[e * KP_ + j] = (__bf16)acc;
  }
}

// ---------------------------------------------------------------------------
// Main: per (b,f) pair, out[b,t,f,e] = sum_j W_eff[e,j]*series[t+j] + B_eff[e]
// as a WMMA GEMM: M=t (512), N=e (128), K=64 (2 x v_wmma_f32_16x16x32_bf16),
// accumulator seeded with the bias.  Grid = 1024 blocks (b*128+f),
// block = 256 threads = 8 wave32.
// ---------------------------------------------------------------------------
__global__ __launch_bounds__(256) void fir_wmma_kernel(
    const float*  __restrict__ x,     // (8,512,128)
    const __bf16* __restrict__ Wbf,   // (128,64)
    const float*  __restrict__ Beff,  // (128,)
    float*        __restrict__ out)   // (8,512,128,128)
{
  __shared__ __bf16 sS[T_ + KP_];     // 576 bf16 = 1152 B

  const int blk = blockIdx.x;
  const int b   = blk >> 7;
  const int f   = blk & (F_ - 1);
  const int tid = threadIdx.x;

  // Build padded time series s[i] = xp2[b, f, i-2]:
  //   xp2 idx <0 or >=543        -> 0        (conv zero-pad)
  //   xp2 idx in [0,15)|[527,543)-> PAD_VAL  (margin pad)
  //   else                       -> x[b, idx-15, f]
  for (int i = tid; i < T_ + KP_; i += 256) {
    int idx = i - 2;
    float v;
    if (idx < 0 || idx >= 543)           v = 0.f;
    else if (idx < 15 || idx >= 15 + T_) v = PADV;
    else                                 v = x[(b * T_ + (idx - 15)) * F_ + f];
    sS[i] = (__bf16)v;
  }
  __syncthreads();

  const int lane  = tid & 31;
  const int wave  = tid >> 5;     // 0..7
  const int nlane = lane & 15;    // N (=e) within 16-wide tile
  const int khalf = lane >> 4;    // 0: low K-half lanes, 1: high

  // B fragments: 8 n-tiles x 2 k-steps, resident in VGPRs for whole kernel.
  // B k order used here (contiguous, matches A): lanes 0-15 hold N=0..15 with K=0..15 in elems 0..15;
  // lanes 16-31 hold same N with K=16..31. Contiguous 32 B -> one v8u load.
  v16bf bfrag[2][8];
  float biasv[8];
  const char* wbytes = (const char*)Wbf;
  #pragma unroll
  for (int nt = 0; nt < 8; ++nt) {
    const int e = nt * 16 + nlane;
    biasv[nt] = Beff[e];
    #pragma unroll
    for (int ks = 0; ks < 2; ++ks) {
      const int kb = ks * 32 + khalf * 16;
      v8u raw = *(const v8u*)(wbytes + (size_t)(e * KP_ + kb) * 2);
      bfrag[ks][nt] = __builtin_bit_cast(v16bf, raw);
    }
  }

  // A k order used here (contiguous, matches B): lane m=lane&15 holds row M=m; lanes 0-15 carry
  // K = 0..15 (elems 0..15); lanes 16-31 carry
  // K = 16..31.  A[t_local, k] = s[t0 + t_local + ks*32 + k].
  const int m     = nlane;
  const int kbase = khalf * 16;

  // Per-lane streaming output pointer. C/D layout: VGPR r -> M-row
  // t = t0 + khalf*8 + r, N-lane -> e = nt*16 + nlane.  After unrolling,
  // r / nt become compile-time byte offsets folded into the store immediate
  // (max 7*65536 + 448 bytes, fits the signed 24-bit IOFFSET).
  const size_t rowStride = (size_t)F_ * E_;   // floats per t-row (16384)
  float* outp = out + ((size_t)b * T_ + (size_t)(wave * 16 + khalf * 8)) * rowStride
                    + (size_t)f * E_ + (size_t)nlane;

  for (int rt = wave; rt < T_ / 16; rt += 8) {
    const int t0 = rt * 16;

    v16bf afrag[2];
    #pragma unroll
    for (int ks = 0; ks < 2; ++ks) {
      const int base = t0 + m + ks * 32 + kbase;
      v16bf av;
      #pragma unroll
      for (int i = 0; i < 8; ++i) av[i]     = sS[base + i];
      #pragma unroll
      for (int i = 0; i < 8; ++i) av[8 + i] = sS[base + 8 + i];
      afrag[ks] = av;
    }

    #pragma unroll
    for (int nt = 0; nt < 8; ++nt) {
      // Seed accumulator with the effective bias (all 8 rows share e).
      v8f c;
      #pragma unroll
      for (int i = 0; i < 8; ++i) c[i] = biasv[nt];

      c = __builtin_amdgcn_wmma_f32_16x16x32_bf16(
              false, afrag[0], false, bfrag[0][nt], (short)0, c, false, false);
      c = __builtin_amdgcn_wmma_f32_16x16x32_bf16(
              false, afrag[1], false, bfrag[1][nt], (short)0, c, false, false);

      // Streaming (non-temporal) stores: output is write-once, never re-read.
      #pragma unroll
      for (int r = 0; r < 8; ++r)
        __builtin_nontemporal_store(c[r], outp + (size_t)r * rowStride + nt * 16);
    }
    outp += rowStride * 128;   // advance 8 row-tiles = 128 t-rows
  }
}

// ---------------------------------------------------------------------------
extern "C" void kernel_launch(void* const* d_in, const int* in_sizes, int n_in,
                              void* d_out, int out_size, void* d_ws, size_t ws_size,
                              hipStream_t stream) {
  (void)in_sizes; (void)n_in; (void)out_size; (void)ws_size;

  const float* x      = (const float*)d_in[0];
  const float* conv_w = (const float*)d_in[1];
  const float* conv_b = (const float*)d_in[2];
  const float* lin_w  = (const float*)d_in[3];
  const float* lin_b  = (const float*)d_in[4];
  float* out = (float*)d_out;

  __bf16* Wbf  = (__bf16*)d_ws;                                   // 16 KB
  float*  Beff = (float*)((char*)d_ws + (size_t)E_ * KP_ * 2);    // +512 B

  prep_kernel<<<1, 128, 0, stream>>>(conv_w, conv_b, lin_w, lin_b, Wbf, Beff);
  fir_wmma_kernel<<<B_ * F_, 256, 0, stream>>>(x, Wbf, Beff, out);
}
